// FCOS_8890582303298
// MI455X (gfx1250) — matrix operand order repaired and checked by reading, not compiled, not run on hardware
//
#include <hip/hip_runtime.h>
#include <hip/hip_bf16.h>

typedef __attribute__((ext_vector_type(8)))  __bf16 v8bf;
typedef __attribute__((ext_vector_type(16))) __bf16 v16bf;
typedef __attribute__((ext_vector_type(8)))  float  v8f;

// ---------------------------------------------------------------------------
// Weight packing: OIHW f32 -> bf16 [tap][ci_blk][co][ci_lo(32)]
// B fragment (32x16 K x N) per lane is then 32 contiguous bytes.
// ---------------------------------------------------------------------------
__global__ void pack_w(const float* __restrict__ w, __bf16* __restrict__ o,
                       int CO_src, int CO_pad) {
    int i = blockIdx.x * blockDim.x + threadIdx.x;
    int tot = CO_src * 256 * 9;
    if (i >= tot) return;
    int t  = i % 9;
    int ci = (i / 9) & 255;
    int co = i / (9 * 256);
    // w flat OIHW: ((co*256 + ci)*3 + kh)*3 + kw == (co*256+ci)*9 + t
    o[((t * 8 + (ci >> 5)) * CO_pad + co) * 32 + (ci & 31)] = (__bf16)w[i];
}

// ---------------------------------------------------------------------------
// Feature convert: NCHW f32 -> halo-padded NHWC bf16  (halo pre-zeroed)
// ---------------------------------------------------------------------------
__global__ void convert_feat(const float* __restrict__ x, __bf16* __restrict__ xp,
                             int H, int W, int HW, int tot) {
    int i = blockIdx.x * blockDim.x + threadIdx.x;
    if (i >= tot) return;   // tot = N*256*HW
    int ww = i % W;
    int hh = (i / W) % H;
    int c  = (i / HW) & 255;
    int n  = i / (HW * 256);
    xp[((n * (H + 2) + hh + 1) * (W + 2) + (ww + 1)) * 256 + c] = (__bf16)x[i];
}

// A fragment: 16 pixels x 32 ci, per-lane two contiguous 16B loads
__device__ __forceinline__ v16bf load_a(const __bf16* __restrict__ xb) {
    v8bf lo = *(const v8bf*)(xb);        // K {0..7}+s
    v8bf hi = *(const v8bf*)(xb + 8);    // K {8..15}+s
    return __builtin_shufflevector(lo, hi,
        0, 1, 2, 3, 4, 5, 6, 7, 8, 9, 10, 11, 12, 13, 14, 15);
}

// ---------------------------------------------------------------------------
// Tower conv: one wave -> 32 pixels x 32 out-channels (2x2 WMMA sub-tiles),
// K = 9 taps x 256 ci. 2 A-frags + 2 B-frags feed 4 WMMAs per k-step
// (halves L0 fragment traffic per FLOP vs a 16x16 tile).
// ---------------------------------------------------------------------------
__global__ __launch_bounds__(128) void conv3x3_tower(
    const __bf16* __restrict__ xp, const __bf16* __restrict__ wp,
    const float* __restrict__ bias, float* __restrict__ y,
    int H, int W, int HW, int total, int co_blocks) {
    const int lane = threadIdx.x & 31;
    const int wave = threadIdx.x >> 5;
    const int cop = blockIdx.y * 4 + wave;      // 32-channel block index
    if (cop >= co_blocks) return;
    const int co0 = cop * 32;
    const int m0 = blockIdx.x * 32;
    const int Wp = W + 2;

    const int s = (lane >= 16) ? 16 : 0;        // A K-half select (contiguous)
    const __bf16* xpix0;
    const __bf16* xpix1;
    {
        int p = m0 + (lane & 15);
        if (p >= total) p = total - 1;
        int n = p / HW, rem = p - n * HW, h = rem / W, w = rem - h * W;
        xpix0 = xp + ((n * (H + 2) + h) * Wp + w) * 256 + s;
        p = m0 + 16 + (lane & 15);
        if (p >= total) p = total - 1;
        n = p / HW; rem = p - n * HW; h = rem / W; w = rem - h * W;
        xpix1 = xp + ((n * (H + 2) + h) * Wp + w) * 256 + s;
    }
    const int bsel = (lane >= 16) ? 16 : 0;     // B K-half select
    const __bf16* wbase = wp + (co0 + (lane & 15)) * 32 + bsel;

    v8f a00 = {0.f,0.f,0.f,0.f,0.f,0.f,0.f,0.f};
    v8f a01 = a00, a10 = a00, a11 = a00;

    for (int t = 0; t < 9; ++t) {
        const int kh = t / 3, kw = t - kh * 3;
        const int aoff = (kh * Wp + kw) * 256;
        const __bf16* wb = wbase + (t * 8) * 256 * 32;
#pragma unroll
        for (int cb = 0; cb < 8; ++cb) {
            v16bf A0 = load_a(xpix0 + aoff + cb * 32);
            v16bf A1 = load_a(xpix1 + aoff + cb * 32);
            const __bf16* wcb = wb + cb * 256 * 32;
            v16bf B0 = *(const v16bf*)(wcb);
            v16bf B1 = *(const v16bf*)(wcb + 16 * 32);
            a00 = __builtin_amdgcn_wmma_f32_16x16x32_bf16(false, A0, false, B0, (short)0, a00, false, false);
            a01 = __builtin_amdgcn_wmma_f32_16x16x32_bf16(false, A0, false, B1, (short)0, a01, false, false);
            a10 = __builtin_amdgcn_wmma_f32_16x16x32_bf16(false, A1, false, B0, (short)0, a10, false, false);
            a11 = __builtin_amdgcn_wmma_f32_16x16x32_bf16(false, A1, false, B1, (short)0, a11, false, false);
        }
    }

    const int nco = co0 + (lane & 15);          // D column
    const int mb = (lane >= 16) ? 8 : 0;        // D rows r / r+8
    const float bv0 = bias[nco];
    const float bv1 = bias[nco + 16];
#pragma unroll
    for (int r = 0; r < 8; ++r) {
        int p0 = m0 + mb + r;
        if (p0 < total) {
            y[p0 * 256 + nco]      = a00[r] + bv0;
            y[p0 * 256 + nco + 16] = a01[r] + bv1;
        }
        int p1 = m0 + 16 + mb + r;
        if (p1 < total) {
            y[p1 * 256 + nco]      = a10[r] + bv0;
            y[p1 * 256 + nco + 16] = a11[r] + bv1;
        }
    }
}

// ---------------------------------------------------------------------------
// Head conv (16x16 tile; <6% of FLOPs): + bias (+optional relu),
// scatter into [N, totHW, Kst] output
// ---------------------------------------------------------------------------
__global__ __launch_bounds__(128) void conv3x3_head(
    const __bf16* __restrict__ xp, const __bf16* __restrict__ wp,
    const float* __restrict__ bias, float* __restrict__ out,
    int H, int W, int HW, int total, int CO_pad, int co_tiles,
    int Kvalid, int Kst, int totHW, int lvl_off, int do_relu) {
    const int lane = threadIdx.x & 31;
    const int wave = threadIdx.x >> 5;
    const int co_tile = blockIdx.y * 4 + wave;
    if (co_tile >= co_tiles) return;
    const int co0 = co_tile * 16;
    const int tileM = blockIdx.x;
    const int Wp = W + 2;

    const int s = (lane >= 16) ? 16 : 0;
    int p = tileM * 16 + (lane & 15);
    if (p >= total) p = total - 1;
    int n = p / HW, rem = p - n * HW, h = rem / W, w = rem - h * W;
    const __bf16* xpix = xp + ((n * (H + 2) + h) * Wp + w) * 256 + s;
    const int bsel = (lane >= 16) ? 16 : 0;
    const int bco  = co0 + (lane & 15);

    v8f acc = {0.f,0.f,0.f,0.f,0.f,0.f,0.f,0.f};
    for (int t = 0; t < 9; ++t) {
        const int kh = t / 3, kw = t - kh * 3;
        const __bf16* xb = xpix + (kh * Wp + kw) * 256;
        const __bf16* wb = wp + ((t * 8) * CO_pad + bco) * 32 + bsel;
#pragma unroll
        for (int cb = 0; cb < 8; ++cb) {
            v16bf a = load_a(xb + cb * 32);
            v16bf b = *(const v16bf*)(wb + cb * CO_pad * 32);
            acc = __builtin_amdgcn_wmma_f32_16x16x32_bf16(
                false, a, false, b, (short)0, acc, false, false);
        }
    }

    const int co = co0 + (lane & 15);
    if (co >= Kvalid) return;
    const float bv = bias[co];
    const int mbase = (lane >= 16) ? 8 : 0;
#pragma unroll
    for (int r = 0; r < 8; ++r) {
        int pp = tileM * 16 + mbase + r;
        if (pp < total) {
            int nn = pp / HW; int hw = pp - nn * HW;
            float v = acc[r] + bv;
            if (do_relu) v = fmaxf(v, 0.f);
            out[(size_t)(nn * totHW + lvl_off + hw) * Kst + co] = v;
        }
    }
}

// ---------------------------------------------------------------------------
// GroupNorm stats: per (n, group=16ch) sum & sumsq via LDS tree + f32 atomics
// ---------------------------------------------------------------------------
__global__ __launch_bounds__(256) void gn_stats(
    const float* __restrict__ y, float* __restrict__ stats,
    int HW, int chunkElems) {
    const int n = blockIdx.z, g = blockIdx.y, chunk = blockIdx.x;
    const int tid = threadIdx.x;
    const int elems = HW * 16;
    int e0 = chunk * chunkElems;
    int e1 = e0 + chunkElems; if (e1 > elems) e1 = elems;

    float a = 0.f, b = 0.f;
    for (int e = e0 + tid; e < e1; e += 256) {
        int sp = e >> 4;
        int c  = (g << 4) + (e & 15);
        float v = y[((size_t)n * HW + sp) * 256 + c];
        a += v; b += v * v;
    }
    __shared__ float s0[256];
    __shared__ float s1[256];
    s0[tid] = a; s1[tid] = b;
    __syncthreads();
#pragma unroll
    for (int o = 128; o > 0; o >>= 1) {
        if (tid < o) { s0[tid] += s0[tid + o]; s1[tid] += s1[tid + o]; }
        __syncthreads();
    }
    if (tid == 0) {
        atomicAdd(&stats[(n * 16 + g) * 2 + 0], s0[0]);
        atomicAdd(&stats[(n * 16 + g) * 2 + 1], s1[0]);
    }
}

// ---------------------------------------------------------------------------
// GroupNorm apply + ReLU -> padded NHWC bf16 (next layer's input)
// ---------------------------------------------------------------------------
__global__ void gn_norm_relu(
    const float* __restrict__ y, const float* __restrict__ stats,
    const float* __restrict__ gamma, const float* __restrict__ beta,
    __bf16* __restrict__ xp, int H, int W, int HW, int total) {
    int i = blockIdx.x * blockDim.x + threadIdx.x;
    if (i >= total * 256) return;
    int c = i & 255;
    int p = i >> 8;
    int n = p / HW; int hw = p - n * HW;
    int h = hw / W; int w = hw - h * W;
    int g = c >> 4;
    float cnt = (float)(HW * 16);
    float mu  = stats[(n * 16 + g) * 2 + 0] / cnt;
    float ex2 = stats[(n * 16 + g) * 2 + 1] / cnt;
    float rs  = rsqrtf(ex2 - mu * mu + 1e-5f);
    float v = (y[i] - mu) * rs * gamma[c] + beta[c];
    v = fmaxf(v, 0.f);
    xp[((size_t)(n * (H + 2) + h + 1) * (W + 2) + (w + 1)) * 256 + c] = (__bf16)v;
}

// ---------------------------------------------------------------------------
// Host orchestration
// ---------------------------------------------------------------------------
static inline void* carve(char** cur, size_t bytes) {
    void* p = (void*)*cur;
    *cur += (bytes + 255) & ~(size_t)255;
    return p;
}

extern "C" void kernel_launch(void* const* d_in, const int* in_sizes, int n_in,
                              void* d_out, int out_size, void* d_ws, size_t ws_size,
                              hipStream_t stream) {
    (void)in_sizes; (void)n_in; (void)out_size; (void)ws_size;
    const int N = 2;
    static const int LH[5] = {100, 50, 25, 13, 7};
    static const int LW[5] = {152, 76, 38, 19, 10};
    int off[5], totHW = 0;
    for (int l = 0; l < 5; ++l) { off[l] = totHW; totHW += LH[l] * LW[l]; }

    // ---- workspace carve ----
    char* cur = (char*)d_ws;
    const size_t WPT_B  = (size_t)72 * 256 * 32 * 2;           // tower weights
    const size_t WPC_B  = (size_t)72 * 80 * 32 * 2;            // cls head
    const size_t WP16_B = (size_t)72 * 16 * 32 * 2;            // reg/ctr head (padded)
    const size_t XP_B   = (size_t)N * 102 * 154 * 256 * 2;     // max padded bf16 act
    const size_t Y_B    = (size_t)N * 100 * 152 * 256 * 4;     // max f32 conv out
    __bf16* wpT[6];
    for (int k = 0; k < 6; ++k) wpT[k] = (__bf16*)carve(&cur, WPT_B);
    __bf16* wpCls = (__bf16*)carve(&cur, WPC_B);
    __bf16* wpReg = (__bf16*)carve(&cur, WP16_B);
    __bf16* wpCtr = (__bf16*)carve(&cur, WP16_B);
    __bf16* xpF = (__bf16*)carve(&cur, XP_B);
    __bf16* xpA = (__bf16*)carve(&cur, XP_B);
    __bf16* xpB = (__bf16*)carve(&cur, XP_B);
    float*  yws   = (float*)carve(&cur, Y_B);
    float*  stats = (float*)carve(&cur, 256);

    // ---- pack weights ----
    {
        int wIdx[6] = {5, 9, 13, 19, 23, 27};   // cw0,cw1,cw2, rw0,rw1,rw2
        int tot = 256 * 256 * 9;
        for (int k = 0; k < 6; ++k)
            pack_w<<<(tot + 255) / 256, 256, 0, stream>>>(
                (const float*)d_in[wIdx[k]], wpT[k], 256, 256);
        int totc = 80 * 256 * 9;
        pack_w<<<(totc + 255) / 256, 256, 0, stream>>>(
            (const float*)d_in[17], wpCls, 80, 80);
        hipMemsetAsync(wpReg, 0, WP16_B, stream);
        hipMemsetAsync(wpCtr, 0, WP16_B, stream);
        int totr = 4 * 256 * 9;
        pack_w<<<(totr + 255) / 256, 256, 0, stream>>>(
            (const float*)d_in[31], wpReg, 4, 16);
        int tott = 1 * 256 * 9;
        pack_w<<<(tott + 255) / 256, 256, 0, stream>>>(
            (const float*)d_in[33], wpCtr, 1, 16);
    }

    float* out_cls = (float*)d_out;
    float* out_reg = out_cls + (size_t)N * totHW * 80;
    float* out_ctr = out_reg + (size_t)N * totHW * 4;

    for (int l = 0; l < 5; ++l) {
        const int H = LH[l], W = LW[l], HW = H * W, total = N * HW;
        const size_t xpBytes = (size_t)N * (H + 2) * (W + 2) * 256 * 2;
        hipMemsetAsync(xpF, 0, xpBytes, stream);  // zero halos
        hipMemsetAsync(xpA, 0, xpBytes, stream);
        hipMemsetAsync(xpB, 0, xpBytes, stream);
        {
            int tot = N * 256 * HW;
            convert_feat<<<(tot + 255) / 256, 256, 0, stream>>>(
                (const float*)d_in[l], xpF, H, W, HW, tot);
        }
        for (int tw = 0; tw < 2; ++tw) {
            const __bf16* in = xpF;
            __bf16* pong[3] = {xpA, xpB, xpA};
            for (int j = 0; j < 3; ++j) {
                const int base = (tw == 0 ? 5 : 19) + j * 4;  // w,b,g,be
                dim3 gc((total + 31) / 32, 2);                // 8 x 32-co blocks / 4 waves
                conv3x3_tower<<<gc, 128, 0, stream>>>(
                    in, wpT[tw * 3 + j], (const float*)d_in[base + 1], yws,
                    H, W, HW, total, 8);
                hipMemsetAsync(stats, 0, 256, stream);
                int chunks = (HW * 16 + 4095) / 4096;
                gn_stats<<<dim3(chunks, 16, N), 256, 0, stream>>>(yws, stats, HW, 4096);
                int ne = total * 256;
                gn_norm_relu<<<(ne + 255) / 256, 256, 0, stream>>>(
                    yws, stats, (const float*)d_in[base + 2],
                    (const float*)d_in[base + 3], pong[j], H, W, HW, total);
                in = pong[j];
            }
            if (tw == 0) {
                dim3 gh((total + 15) / 16, 2);   // ceil(5 co-tiles / 4 waves)
                conv3x3_head<<<gh, 128, 0, stream>>>(
                    in, wpCls, (const float*)d_in[18], out_cls,
                    H, W, HW, total, 80, 5, 80, 80, totHW, off[l], 0);
            } else {
                dim3 gh((total + 15) / 16, 1);
                conv3x3_head<<<gh, 128, 0, stream>>>(
                    in, wpReg, (const float*)d_in[32], out_reg,
                    H, W, HW, total, 16, 1, 4, 4, totHW, off[l], 1);
                conv3x3_head<<<gh, 128, 0, stream>>>(
                    in, wpCtr, (const float*)d_in[34], out_ctr,
                    H, W, HW, total, 16, 1, 1, 1, totHW, off[l], 0);
            }
        }
    }
}
